// MultiHeadSelfAttention_79611513799381
// MI455X (gfx1250) — matrix operand order repaired and checked by reading, not compiled, not run on hardware
//
#include <hip/hip_runtime.h>
#include <hip/hip_bf16.h>

#define D_MODEL 1024
#define SEQ     2048
#define BATCH   4
#define NHEAD   16
#define HSZ     64
#define MROWS   (BATCH*SEQ)      // 8192
#define MT      (MROWS/16)       // 512 16-row tiles
#define MT2     (MROWS/32)       // 256 32-row tiles
#define KT32    (D_MODEL/32)     // 32 k tiles
#define NC      (SEQ/32)         // 64 attention k-chunks

typedef __attribute__((ext_vector_type(16))) _Float16 v16h;
typedef __attribute__((ext_vector_type(8)))  _Float16 v8h;
typedef __attribute__((ext_vector_type(8)))  float    v8f;

union V16U { v16h v; v8h h[2]; };

static __device__ inline v8f zero8() {
  v8f z;
#pragma unroll
  for (int i = 0; i < 8; ++i) z[i] = 0.f;
  return z;
}

// Wave-local LDS fence: DS ops are in-order per wave, so a dscnt drain plus a
// compiler memory barrier orders our store->load LDS reuse without a full
// workgroup barrier (waves in a block touch disjoint LDS slices here).
#define LDS_FENCE() asm volatile("s_wait_dscnt 0x0" ::: "memory")

// f16 element counts for workspace layout
#define XP_ELEMS  ((size_t)MT * KT32 * 32 * 16)        // 8,388,608  (16 MB)
#define WP_ELEMS  ((size_t)NHEAD * 4 * KT32 * 32 * 16) // 1,048,576  (2 MB each)
#define QKV_ELEMS ((size_t)BATCH * NHEAD * SEQ * HSZ)  // 8,388,608  (16 MB each)

// ---------------------------------------------------------------------------
// Pack X (fp32 row-major [8192][1024]) into f16 WMMA A-fragment order:
// index = ((mT*32 + kT)*32 + lane)*16 + e
// ---------------------------------------------------------------------------
__global__ __launch_bounds__(256) void pack_x_kernel(const float* __restrict__ x,
                                                     _Float16* __restrict__ xp) {
  int gid  = blockIdx.x * 256 + threadIdx.x;   // 0 .. MT*KT32*32-1
  int lane = gid & 31;
  int t    = gid >> 5;
  int kT   = t & (KT32 - 1);
  int mT   = t >> 5;
  int row  = lane & 15;
  int k0   = kT * 32 + ((lane & 16) ? 16 : 0);
  const float* src = x + (size_t)(mT * 16 + row) * D_MODEL + k0;
  _Float16* dst = xp + (size_t)gid * 16;
#pragma unroll
  for (int e = 0; e < 8; ++e) dst[e] = (_Float16)src[e];
#pragma unroll
  for (int e = 0; e < 8; ++e) dst[8 + e] = (_Float16)src[8 + e];
}

// ---------------------------------------------------------------------------
// Pack W (fp32 row-major [1024][1024]) into f16 WMMA B-fragment order:
// index = (((nb*4+nt)*32 + kT)*32 + lane)*16 + e
// ---------------------------------------------------------------------------
__global__ __launch_bounds__(256) void pack_w_kernel(const float* __restrict__ wq,
                                                     const float* __restrict__ wk,
                                                     const float* __restrict__ wv,
                                                     _Float16* __restrict__ wp) {
  int sel = blockIdx.y;
  const float* w = (sel == 0) ? wq : ((sel == 1) ? wk : wv);
  _Float16* base = wp + (size_t)sel * WP_ELEMS;

  int gid   = blockIdx.x * 256 + threadIdx.x;  // 0 .. NHEAD*4*KT32*32-1
  int lane  = gid & 31;
  int t     = gid >> 5;
  int kT    = t & (KT32 - 1);
  int nbnt  = t >> 5;                          // nb*4 + nt
  int col   = lane & 15;
  int rbase = (lane & 16) ? 16 : 0;
  int n     = nbnt * 16 + col;
  const float* src = w + (size_t)(kT * 32 + rbase) * D_MODEL + n;
  _Float16* dst = base + (size_t)gid * 16;
#pragma unroll
  for (int e = 0; e < 16; ++e) dst[e] = (_Float16)src[(size_t)e * D_MODEL];
}

// ---------------------------------------------------------------------------
// QKV projection GEMM: one wave computes a 32x64 strip. Two-slot ping-pong
// software pipeline (no register rotation copies): compute(slot0,k) ->
// refill(slot0,k+2) -> compute(slot1,k+1) -> refill(slot1,k+3).
// Q,K stored [b,h,s,d] f16 row-major; V stored transposed [b,h,d,s] f16.
// ---------------------------------------------------------------------------
__global__ __launch_bounds__(128) void qkv_gemm_kernel(
    const _Float16* __restrict__ xp, const _Float16* __restrict__ wp,
    const float* __restrict__ bq, const float* __restrict__ bk,
    const float* __restrict__ bv,
    _Float16* __restrict__ qb, _Float16* __restrict__ kb,
    _Float16* __restrict__ vt) {
  int sel = blockIdx.z;
  const _Float16* wsel = wp + (size_t)sel * WP_ELEMS;
  const float* bias = (sel == 0) ? bq : ((sel == 1) ? bk : bv);

  int wave = threadIdx.x >> 5, lane = threadIdx.x & 31;
  int m2 = blockIdx.x * 4 + wave;  // 0..255 -> rows m2*32 .. +31
  int nb = blockIdx.y;             // head

  const v16h* A0 = (const v16h*)xp + (size_t)(2 * m2) * KT32 * 32 + lane;
  const v16h* A1 = A0 + (size_t)KT32 * 32;
  const v16h* Bp = (const v16h*)wsel + (size_t)nb * 4 * KT32 * 32 + lane;

  v8f acc[2][4];
#pragma unroll
  for (int t = 0; t < 2; ++t)
#pragma unroll
    for (int nt = 0; nt < 4; ++nt) acc[t][nt] = zero8();

  v16h a0s[2], a1s[2], bs[2][4];   // ping-pong fragment slots

  auto refill = [&](int slot, int k) {
    a0s[slot] = A0[(size_t)k * 32];
    a1s[slot] = A1[(size_t)k * 32];
#pragma unroll
    for (int nt = 0; nt < 4; ++nt) bs[slot][nt] = Bp[((size_t)nt * KT32 + k) * 32];
  };
  auto do8 = [&](int slot) {
#pragma unroll
    for (int nt = 0; nt < 4; ++nt)
      acc[0][nt] = __builtin_amdgcn_wmma_f32_16x16x32_f16(
          false, a0s[slot], false, bs[slot][nt], (short)0, acc[0][nt], false, false);
#pragma unroll
    for (int nt = 0; nt < 4; ++nt)
      acc[1][nt] = __builtin_amdgcn_wmma_f32_16x16x32_f16(
          false, a1s[slot], false, bs[slot][nt], (short)0, acc[1][nt], false, false);
  };

  refill(0, 0);
  refill(1, 1);
  for (int kT = 0; kT < KT32 - 2; kT += 2) {
    do8(0);
    refill(0, kT + 2);    // overlaps do8(1)
    do8(1);
    refill(1, kT + 3);    // overlaps next do8(0)
  }
  do8(0);                 // k = 30
  do8(1);                 // k = 31

  int col = lane & 15;
  int rlo = (lane & 16) ? 8 : 0;
#pragma unroll
  for (int t = 0; t < 2; ++t) {
    int mT = 2 * m2 + t;
#pragma unroll
    for (int nt = 0; nt < 4; ++nt) {
#pragma unroll
      for (int r = 0; r < 8; ++r) {
        int m_abs = mT * 16 + rlo + r;
        int bb = m_abs >> 11;           // /SEQ
        int s  = m_abs & (SEQ - 1);
        int d  = nt * 16 + col;
        float val = acc[t][nt][r] + bias[nb * 64 + d];
        if (sel == 0)
          qb[((size_t)(bb * NHEAD + nb) * SEQ + s) * HSZ + d] = (_Float16)val;
        else if (sel == 1)
          kb[((size_t)(bb * NHEAD + nb) * SEQ + s) * HSZ + d] = (_Float16)val;
        else
          vt[((size_t)(bb * NHEAD + nb) * HSZ + d) * SEQ + s] = (_Float16)val;
      }
    }
  }
}

// ---------------------------------------------------------------------------
// Flash-attention: block = 4 waves, each wave owns 16 q rows of one (b,h).
// K fragments are double-buffered across k-chunks (next chunk's K loads are
// issued right after the current score WMMAs, hiding latency behind softmax
// and PV). V loads issue at the top of each chunk and are consumed at the end.
// Softmax split across all 32 lanes; halves combined with __shfl_xor(,16).
// Wave-local s_wait_dscnt fences instead of workgroup barriers.
// ---------------------------------------------------------------------------
__global__ __launch_bounds__(128) void attn_kernel(const _Float16* __restrict__ Q,
                                                   const _Float16* __restrict__ K,
                                                   const _Float16* __restrict__ Vt,
                                                   float* __restrict__ out) {
  __shared__ float sS[4][16][33];                 // score tile per wave (padded)
  __shared__ __align__(32) _Float16 sP[4][16][32];
  __shared__ float sAlpha[4][16];
  __shared__ float sL[4][16];

  int wave = threadIdx.x >> 5, lane = threadIdx.x & 31;
  int bh = blockIdx.y;
  int b = bh >> 4, h = bh & 15;
  int q0 = blockIdx.x * 64 + wave * 16;

  const _Float16* Qb = Q  + (size_t)bh * SEQ * HSZ;
  const _Float16* Kb = K  + (size_t)bh * SEQ * HSZ;
  const _Float16* Vb = Vt + (size_t)bh * HSZ * SEQ;

  int arow  = lane & 15;
  int aoff  = (lane & 16) ? 16 : 0;  // A-fragment element offset
  int brow  = lane & 15;             // B-fragment column (n)
  int bbase = (lane & 16) ? 16 : 0;  // B-fragment K-row base
  int col   = lane & 15;
  int rlo   = (lane & 16) ? 8 : 0;
  int half  = lane >> 4;             // softmax column-half owned by this lane

  // Q A-fragments (reused across the whole k loop)
  v16h qf[2];
#pragma unroll
  for (int dk = 0; dk < 2; ++dk) {
    const _Float16* p = Qb + (size_t)(q0 + arow) * HSZ + dk * 32 + aoff;
    V16U u;
    u.h[0] = *(const v8h*)p;
    u.h[1] = *(const v8h*)(p + 8);
    qf[dk] = u.v;
  }

  v8f o[4];
#pragma unroll
  for (int nt = 0; nt < 4; ++nt) o[nt] = zero8();
  float m_i = -3.0e38f, l_i = 0.f;   // maintained consistently in all lanes

  v16h kfb[2][2][2];                 // [stage][tile][dk] K fragments

  auto loadK = [&](int st, int kc) {
    int ss = kc * 32;
#pragma unroll
    for (int t = 0; t < 2; ++t)
#pragma unroll
      for (int dk = 0; dk < 2; ++dk)
        kfb[st][t][dk] = *(const v16h*)(Kb + (size_t)(ss + t * 16 + brow) * HSZ +
                                        dk * 32 + bbase);
  };

  auto step = [&](int st, int kc) {
    int s0 = kc * 32;

    // ---- V fragments for this chunk (consumed at the end) ----
    v16h vf[4];
#pragma unroll
    for (int nt = 0; nt < 4; ++nt)
      vf[nt] = *(const v16h*)(Vb + (size_t)(nt * 16 + brow) * SEQ + s0 + bbase);

    // ---- scores: two 16x16 tiles, K-dim = d (64 = 2x32) ----
    v8f sc[2];
#pragma unroll
    for (int t = 0; t < 2; ++t) {
      sc[t] = zero8();
#pragma unroll
      for (int dk = 0; dk < 2; ++dk)
        sc[t] = __builtin_amdgcn_wmma_f32_16x16x32_f16(
            false, qf[dk], false, kfb[st][t][dk], (short)0, sc[t], false, false);
    }

    // ---- issue next chunk's K loads now: hidden behind softmax + PV ----
    if (kc + 1 < NC) loadK(st ^ 1, kc + 1);
    if (s0 + 64 < SEQ) {  // prefetch hints two chunks ahead
      __builtin_prefetch(Kb + (size_t)(s0 + 64 + brow) * HSZ, 0, 0);
      __builtin_prefetch(Vb + (size_t)brow * SEQ + s0 + 64, 0, 0);
    }

#pragma unroll
    for (int t = 0; t < 2; ++t)
#pragma unroll
      for (int r = 0; r < 8; ++r)
        sS[wave][rlo + r][t * 16 + col] = sc[t][r] * 0.125f;  // 1/sqrt(64)
    LDS_FENCE();

    // ---- online softmax: each lane scans 16 cols of row (lane&15) ----
    {
      int row = arow;
      int c0 = half * 16;
      float pm = -3.0e38f;
#pragma unroll
      for (int c = 0; c < 16; ++c) pm = fmaxf(pm, sS[wave][row][c0 + c]);
      float om = fmaxf(pm, __shfl_xor(pm, 16));
      float mnew = fmaxf(m_i, om);
      float alpha = __expf(m_i - mnew);
      float ps = 0.f;
#pragma unroll
      for (int c = 0; c < 16; ++c) {
        float p = __expf(sS[wave][row][c0 + c] - mnew);
        ps += p;
        sP[wave][row][c0 + c] = (_Float16)p;
      }
      float ts = ps + __shfl_xor(ps, 16);
      l_i = l_i * alpha + ts;
      m_i = mnew;
      if (lane < 16) sAlpha[wave][row] = alpha;
    }
    LDS_FENCE();

    // ---- rescale running output accumulators ----
#pragma unroll
    for (int r = 0; r < 8; ++r) {
      float ar = sAlpha[wave][rlo + r];
#pragma unroll
      for (int nt = 0; nt < 4; ++nt) o[nt][r] *= ar;
    }

    // ---- P A-fragment from LDS, then P @ V ----
    V16U pu;
    pu.h[0] = *(const v8h*)&sP[wave][arow][aoff];
    pu.h[1] = *(const v8h*)&sP[wave][arow][aoff + 8];
    v16h pf = pu.v;
#pragma unroll
    for (int nt = 0; nt < 4; ++nt)
      o[nt] = __builtin_amdgcn_wmma_f32_16x16x32_f16(
          false, pf, false, vf[nt], (short)0, o[nt], false, false);
    LDS_FENCE();
  };

  loadK(0, 0);
  for (int kc = 0; kc < NC; kc += 2) {   // NC is even; stages alternate 0/1
    step(0, kc);
    step(1, kc + 1);
  }

  if (lane < 16) sL[wave][lane] = 1.f / l_i;
  LDS_FENCE();

  // out[b][s][h*64+d], fp32
#pragma unroll
  for (int nt = 0; nt < 4; ++nt) {
#pragma unroll
    for (int r = 0; r < 8; ++r) {
      int row = rlo + r;
      int s = q0 + row;
      int d = nt * 16 + col;
      out[((size_t)b * SEQ + s) * D_MODEL + h * HSZ + d] = o[nt][r] * sL[wave][row];
    }
  }
}

// ---------------------------------------------------------------------------
extern "C" void kernel_launch(void* const* d_in, const int* in_sizes, int n_in,
                              void* d_out, int out_size, void* d_ws, size_t ws_size,
                              hipStream_t stream) {
  const float* x  = (const float*)d_in[0];
  const float* Wq = (const float*)d_in[1];
  const float* bq = (const float*)d_in[2];
  const float* Wk = (const float*)d_in[3];
  const float* bk = (const float*)d_in[4];
  const float* Wv = (const float*)d_in[5];
  const float* bv = (const float*)d_in[6];
  float* out = (float*)d_out;

  _Float16* xp = (_Float16*)d_ws;
  _Float16* wp = xp + XP_ELEMS;
  _Float16* qb = wp + 3 * WP_ELEMS;
  _Float16* kb = qb + QKV_ELEMS;
  _Float16* vt = kb + QKV_ELEMS;

  pack_x_kernel<<<(MT * KT32 * 32) / 256, 256, 0, stream>>>(x, xp);
  pack_w_kernel<<<dim3((NHEAD * 4 * KT32 * 32) / 256, 3), 256, 0, stream>>>(Wq, Wk, Wv, wp);
  qkv_gemm_kernel<<<dim3(MT2 / 4, NHEAD, 3), 128, 0, stream>>>(
      xp, wp, bq, bk, bv, qb, kb, vt);
  attn_kernel<<<dim3(SEQ / 64, BATCH * NHEAD), 128, 0, stream>>>(qb, kb, vt, out);
}
